// HypAgg_62852551410107
// MI455X (gfx1250) — matrix operand order repaired and checked by reading, not compiled, not run on hardware
//
#include <hip/hip_runtime.h>
#include <cstddef>
#include <cstdint>

// ---------------- problem constants (from reference) ----------------
#define Bb     2
#define Nn     768
#define Dd     64
#define TILE   64
#define PITCH  72          // LDS row pitch in halves (144B rows -> conflict-free b128 loads)
#define NJT    12          // N / TILE
#define NIT    12
#define JSPLIT 4           // j-tiles split across workgroups -> 2*12*4 = 96 WGs
#define MINN   1e-15f
#define MAXNRM 0.996f      // (1 - BALL_EPS)/sqrt(c)
#define CLAMPV 1000000.0f

typedef __attribute__((ext_vector_type(16))) _Float16 v16h;
typedef __attribute__((ext_vector_type(8)))  _Float16 v8h;
typedef __attribute__((ext_vector_type(8)))  float    v8f;

// fast 1-ulp hardware approximations (v_rcp_f32 / v_sqrt_f32)
__device__ inline float frcp(float x)  { return __builtin_amdgcn_rcpf(x); }
__device__ inline float fsqrt(float x) { return __builtin_amdgcn_sqrtf(x); }

// ---------------- CDNA5 async global->LDS DMA (ASYNCcnt-tracked) ----------------
__device__ inline void async_b128(unsigned lds_addr, const float* g) {
    unsigned long long ga = (unsigned long long)(uintptr_t)g;
    asm volatile("global_load_async_to_lds_b128 %0, %1, off"
                 :: "v"(lds_addr), "v"(ga) : "memory");
}
__device__ inline void wait_async0() {
    asm volatile("s_wait_asynccnt 0x0" ::: "memory");
}

// ---------------- WMMA helpers ----------------
__device__ inline v8f wmma16(v16h a, v16h b, v8f c) {
    return __builtin_amdgcn_wmma_f32_16x16x32_f16(
        false, a, false, b, (short)0, c, false, false);
}

// A-fragment: lane holds row (lane&15); lanes 0-15 carry K={0..7,16..23},
// lanes 16-31 carry K={8..15,24..31} of the 32-wide K chunk.
__device__ inline v16h frag_a(const _Float16* t, int row0, int kchunk, int lane) {
    int lr = lane & 15, lh = lane >> 4;
    const _Float16* p = t + (row0 + lr) * PITCH + kchunk * 32 + lh * 8;
    v16h f;
    ((v8h*)&f)[0] = *(const v8h*)p;
    ((v8h*)&f)[1] = *(const v8h*)(p + 16);
    return f;
}

// B-fragment: lane holds column (lane&15); lanes 0-15 carry K={0..7,16..23},
// lanes 16-31 carry K={8..15,24..31} (column n contiguous when tile stored K-major).
__device__ inline v16h frag_b(const _Float16* t, int row0, int kchunk, int lane) {
    int lc = lane & 15, lh = lane >> 4;
    const _Float16* p = t + (row0 + lc) * PITCH + kchunk * 32 + lh * 8;
    v16h f;
    ((v8h*)&f)[0] = *(const v8h*)p;
    ((v8h*)&f)[1] = *(const v8h*)(p + 16);
    return f;
}

// ---------------- kernel 1: per-node scalars ----------------
__global__ void __launch_bounds__(256) hyp_pre(const float* __restrict__ x,
                                               const float* __restrict__ W,
                                               const float* __restrict__ bias,
                                               float* __restrict__ n2g,
                                               float* __restrict__ dw2g,
                                               float* __restrict__ sdg) {
    int idx = blockIdx.x * blockDim.x + threadIdx.x;
    if (idx >= Bb * Nn) return;
    const float* xp = x + (size_t)idx * Dd;
    float n2 = 0.f, d1 = 0.f, d2 = 0.f;
#pragma unroll 8
    for (int d = 0; d < Dd; ++d) {
        float v = xp[d];
        n2 += v * v;
        d1 += v * W[d];
        d2 += v * W[Dd + d];
    }
    n2g[idx]  = n2;
    dw2g[idx] = d2;
    // x_self = logmap(x,x): sub = s*x with s = (beta - alpha)/den (fp cancellation residue)
    float xy  = -n2;
    float al  = 1.f + 2.f * xy + n2;
    float be  = 1.f - n2;
    float den = fmaxf(1.f + 2.f * xy + n2 * n2, MINN);
    float s   = (be - al) * frcp(den);
    float sn  = fmaxf(fabsf(s) * fsqrt(n2), MINN);
    float z   = fminf(sn, 1.f - 1e-7f);
    float at  = 0.5f * __logf((1.f + z) * frcp(1.f - z));
    float g   = fmaxf(be, MINN) * at * frcp(sn);
    sdg[idx]  = g * s * d1 + bias[0];
}

// ---------------- elementwise stage on a WMMA accumulator ----------------
__device__ inline void elemwise(const v8f& G, float* cA, int isub, int jsub,
                                int lane, int bb, int i0, int j0,
                                const float* __restrict__ adj,
                                const float* sN2i, const float* sDW2i, const float* sSd,
                                const float* sN2j, const float* sDW2j,
                                _Float16* sM) {
    int lc = lane & 15, lh = lane >> 4;
    int jl = jsub * 16 + lc;
    float n2j = sN2j[jl], dw2j = sDW2j[jl];
#pragma unroll
    for (int v = 0; v < 8; ++v) {
        int il = isub * 16 + v + lh * 8;
        float n2i  = sN2i[il];
        float dw2i = sDW2i[il];
        float sd   = sSd[il];
        float Gv   = G[v];
        float den  = fmaxf(1.f - 2.f * Gv + n2i * n2j, MINN);
        float rden = frcp(den);
        float Ac   = -(1.f - 2.f * Gv + n2j) * rden;
        float Bc   = (1.f - n2i) * rden;
        float sn2  = Ac * Ac * n2i + 2.f * Ac * Bc * Gv + Bc * Bc * n2j;
        float sn   = fmaxf(fsqrt(fmaxf(sn2, 0.f)), MINN);
        float z    = fminf(sn, 1.f - 1e-7f);
        float at   = 0.5f * __logf((1.f + z) * frcp(1.f - z));
        float g    = fmaxf(1.f - n2i, MINN) * at * frcp(sn);
        float logit = sd + g * (Ac * dw2i + Bc * dw2j);
        float sig  = frcp(1.f + __expf(-logit));
        float a    = adj[((size_t)bb * Nn + (i0 + il)) * (size_t)Nn + (j0 + jl)];
        float w    = sig * a * g;
        cA[v]     += w * Ac;                       // scalar coef on x_i
        sM[il * PITCH + jl] = (_Float16)(w * Bc);  // GEMM2 A-matrix
    }
}

// ---------------- kernel 2: fused Gram -> attention -> aggregate ----------------
__global__ void __launch_bounds__(256) hyp_main(const float* __restrict__ x,
                                                const float* __restrict__ adj,
                                                const float* __restrict__ n2g,
                                                const float* __restrict__ dw2g,
                                                const float* __restrict__ sdg,
                                                float* __restrict__ coefg,
                                                float* __restrict__ Sg) {
    __shared__ float    sStage[TILE * Dd];   // 16 KB f32 DMA staging (thread-private 64B slices)
    __shared__ _Float16 sXi[TILE * PITCH];
    __shared__ _Float16 sXj[TILE * PITCH];
    __shared__ _Float16 sXjT[TILE * PITCH];
    __shared__ _Float16 sM[TILE * PITCH];
    __shared__ float sN2i[TILE], sDW2i[TILE], sSd[TILE];
    __shared__ float sN2j[TILE], sDW2j[TILE];
    __shared__ float sCoef[TILE];

    int wg  = blockIdx.x;
    int bb  = wg / (NIT * JSPLIT);
    int rem = wg % (NIT * JSPLIT);
    int it  = rem / JSPLIT;
    int js  = rem % JSPLIT;
    int i0  = it * TILE;

    int tid = threadIdx.x, lane = tid & 31, wv = tid >> 5;
    const float* xb = x + (size_t)bb * Nn * Dd;

    // each thread owns a fixed 16-float slice: (row = tid/4, cols = (tid&3)*16 ..+15)
    int srow = tid >> 2, scg = (tid & 3) * 16;
    float*    stg   = sStage + srow * Dd + scg;
    unsigned  stgL  = (unsigned)(uintptr_t)(const void*)stg;

    // ---- stage Xi tile via async DMA, convert to f16 ----
#pragma unroll
    for (int k = 0; k < 4; ++k)
        async_b128(stgL + 16u * k, xb + (size_t)(i0 + srow) * Dd + scg + 4 * k);
    wait_async0();
    {
        _Float16* dst = sXi + srow * PITCH + scg;
#pragma unroll
        for (int k = 0; k < 16; ++k) dst[k] = (_Float16)stg[k];
    }
    if (tid < TILE) {
        sN2i[tid]  = n2g[bb * Nn + i0 + tid];
        sDW2i[tid] = dw2g[bb * Nn + i0 + tid];
        sSd[tid]   = sdg[bb * Nn + i0 + tid];
        sCoef[tid] = 0.f;
    }
    // prime the pipeline: DMA the first j-tile
#pragma unroll
    for (int k = 0; k < 4; ++k)
        async_b128(stgL + 16u * k, xb + (size_t)(js * TILE + srow) * Dd + scg + 4 * k);
    __syncthreads();

    int s1 = wv, s2 = wv + 8;
    int is1 = s1 >> 2, js1 = s1 & 3;
    int is2 = s2 >> 2, js2 = s2 & 3;

    v8f accS1 = {0.f, 0.f, 0.f, 0.f, 0.f, 0.f, 0.f, 0.f};
    v8f accS2 = {0.f, 0.f, 0.f, 0.f, 0.f, 0.f, 0.f, 0.f};
    float cA1[8] = {0.f, 0.f, 0.f, 0.f, 0.f, 0.f, 0.f, 0.f};
    float cA2[8] = {0.f, 0.f, 0.f, 0.f, 0.f, 0.f, 0.f, 0.f};

    for (int jt = js; jt < NJT; jt += JSPLIT) {
        int j0 = jt * TILE;

        // ---- consume DMA'd f32 tile: convert to f16 row-major + d-major copies ----
        wait_async0();
        {
            _Float16 h[16];
#pragma unroll
            for (int k = 0; k < 16; ++k) h[k] = (_Float16)stg[k];
            _Float16* dst = sXj + srow * PITCH + scg;
#pragma unroll
            for (int k = 0; k < 16; ++k) dst[k] = h[k];
#pragma unroll
            for (int k = 0; k < 16; ++k) sXjT[(scg + k) * PITCH + srow] = h[k];
        }
        if (tid < TILE) {
            sN2j[tid]  = n2g[bb * Nn + j0 + tid];
            sDW2j[tid] = dw2g[bb * Nn + j0 + tid];
        }
        __syncthreads();

        // ---- software pipeline: kick off next tile's DMA; overlaps both GEMMs ----
        if (jt + JSPLIT < NJT) {
            int jn = (jt + JSPLIT) * TILE;
#pragma unroll
            for (int k = 0; k < 4; ++k)
                async_b128(stgL + 16u * k, xb + (size_t)(jn + srow) * Dd + scg + 4 * k);
        }

        // ---- GEMM1: Gram sub-tiles  G = Xi * Xj^T  (K = 64 -> 2 WMMAs) ----
        v8f G1 = {0.f, 0.f, 0.f, 0.f, 0.f, 0.f, 0.f, 0.f};
        v8f G2 = {0.f, 0.f, 0.f, 0.f, 0.f, 0.f, 0.f, 0.f};
        G1 = wmma16(frag_a(sXi, is1 * 16, 0, lane), frag_b(sXj, js1 * 16, 0, lane), G1);
        G1 = wmma16(frag_a(sXi, is1 * 16, 1, lane), frag_b(sXj, js1 * 16, 1, lane), G1);
        G2 = wmma16(frag_a(sXi, is2 * 16, 0, lane), frag_b(sXj, js2 * 16, 0, lane), G2);
        G2 = wmma16(frag_a(sXi, is2 * 16, 1, lane), frag_b(sXj, js2 * 16, 1, lane), G2);

        // ---- hyperbolic attention directly on the WMMA accumulators ----
        elemwise(G1, cA1, is1, js1, lane, bb, i0, j0, adj, sN2i, sDW2i, sSd, sN2j, sDW2j, sM);
        elemwise(G2, cA2, is2, js2, lane, bb, i0, j0, adj, sN2i, sDW2i, sSd, sN2j, sDW2j, sM);
        __syncthreads();

        // ---- GEMM2: S += M * Xj ----
        accS1 = wmma16(frag_a(sM, is1 * 16, 0, lane), frag_b(sXjT, js1 * 16, 0, lane), accS1);
        accS1 = wmma16(frag_a(sM, is1 * 16, 1, lane), frag_b(sXjT, js1 * 16, 1, lane), accS1);
        accS2 = wmma16(frag_a(sM, is2 * 16, 0, lane), frag_b(sXjT, js2 * 16, 0, lane), accS2);
        accS2 = wmma16(frag_a(sM, is2 * 16, 1, lane), frag_b(sXjT, js2 * 16, 1, lane), accS2);
        __syncthreads();  // protect sXj/sXjT/sM/sN2j for next iteration
    }

    // ---- flush per-row coef partials: regs -> LDS (ds_add_f32) -> global atomic ----
    {
        int lh = lane >> 4;
#pragma unroll
        for (int v = 0; v < 8; ++v) {
            atomicAdd(&sCoef[is1 * 16 + v + lh * 8], cA1[v]);
            atomicAdd(&sCoef[is2 * 16 + v + lh * 8], cA2[v]);
        }
    }
    __syncthreads();
    if (tid < TILE) atomicAdd(&coefg[bb * Nn + i0 + tid], sCoef[tid]);

    // ---- flush S partials via global_atomic_add_f32 ----
    {
        int lc = lane & 15, lh = lane >> 4;
#pragma unroll
        for (int v = 0; v < 8; ++v) {
            int ig1 = i0 + is1 * 16 + v + lh * 8;
            int ig2 = i0 + is2 * 16 + v + lh * 8;
            atomicAdd(&Sg[((size_t)bb * Nn + ig1) * Dd + js1 * 16 + lc], accS1[v]);
            atomicAdd(&Sg[((size_t)bb * Nn + ig2) * Dd + js2 * 16 + lc], accS2[v]);
        }
    }
}

// ---------------- kernel 3: support -> clip -> expmap -> proj ----------------
__global__ void __launch_bounds__(128) hyp_fin(const float* __restrict__ x,
                                               const float* __restrict__ Sg,
                                               const float* __restrict__ coefg,
                                               const float* __restrict__ n2g,
                                               float* __restrict__ out) {
    int idx = blockIdx.x * blockDim.x + threadIdx.x;
    if (idx >= Bb * Nn) return;
    const float* xp = x + (size_t)idx * Dd;
    const float* Sp = Sg + (size_t)idx * Dd;
    float coef = coefg[idx];
    float n2   = n2g[idx];

    float suu = 0.f, sxu = 0.f;
#pragma unroll 8
    for (int d = 0; d < Dd; ++d) {
        float uv = fminf(fmaxf(coef * xp[d] + Sp[d], -CLAMPV), CLAMPV);
        suu += uv * uv;
        sxu += xp[d] * uv;
    }
    float un  = fmaxf(fsqrt(suu), MINN);
    float lam = 2.f * frcp(fmaxf(1.f - n2, MINN));
    float t   = tanhf(0.5f * lam * un);
    float sc  = t * frcp(un);
    float y2  = t * t;
    float xy  = sc * sxu;
    float al  = 1.f + 2.f * xy + y2;
    float be  = 1.f - n2;
    float inv = frcp(fmaxf(1.f + 2.f * xy + n2 * y2, MINN));

    float r2 = 0.f;
    float* op = out + (size_t)idx * Dd;
#pragma unroll 8
    for (int d = 0; d < Dd; ++d) {
        float uv = fminf(fmaxf(coef * xp[d] + Sp[d], -CLAMPV), CLAMPV);
        float r  = (al * xp[d] + be * sc * uv) * inv;
        op[d] = r;
        r2 += r * r;
    }
    float nr = fmaxf(fsqrt(r2), MINN);
    if (nr > MAXNRM) {
        float s = MAXNRM * frcp(nr);
#pragma unroll 8
        for (int d = 0; d < Dd; ++d) op[d] *= s;
    }
}

// ---------------- host-side launcher ----------------
extern "C" void kernel_launch(void* const* d_in, const int* in_sizes, int n_in,
                              void* d_out, int out_size, void* d_ws, size_t ws_size,
                              hipStream_t stream) {
    const float* x    = (const float*)d_in[0];
    const float* adj  = (const float*)d_in[1];
    const float* W    = (const float*)d_in[2];
    const float* bias = (const float*)d_in[3];
    float* out = (float*)d_out;
    float* ws  = (float*)d_ws;

    const int BN = Bb * Nn;
    float* n2g   = ws;
    float* dw2g  = ws + BN;
    float* sdg   = ws + 2 * BN;
    float* coefg = ws + 3 * BN;
    float* Sg    = ws + 4 * BN;

    size_t zbytes = (size_t)(4 * BN + BN * Dd) * sizeof(float);
    hipMemsetAsync(d_ws, 0, zbytes, stream);

    hyp_pre<<<(BN + 255) / 256, 256, 0, stream>>>(x, W, bias, n2g, dw2g, sdg);
    hyp_main<<<Bb * NIT * JSPLIT, 256, 0, stream>>>(x, adj, n2g, dw2g, sdg, coefg, Sg);
    hyp_fin<<<(BN + 127) / 128, 128, 0, stream>>>(x, Sg, coefg, n2g, out);
}
